// MultiHeadSelfAttention_7610682049156
// MI455X (gfx1250) — matrix operand order repaired and checked by reading, not compiled, not run on hardware
//
#include <hip/hip_runtime.h>
#include <hip/hip_bf16.h>

// MultiHeadSelfAttention for MI455X (gfx1250), bf16 WMMA pipeline.
// B=2, S=2048, H=768, NH=12, HD=64.
// Round 2: TDM (tensor_load_to_lds) staging of K/V chunks in attention,
//          32x64 wave tiles in the projection GEMMs.

#define NH 12
#define BB 2
#define SS 2048
#define HH 768
#define HD 64

typedef __bf16 bf16_t;
typedef __attribute__((ext_vector_type(16))) __bf16 v16bf;
typedef __attribute__((ext_vector_type(8)))  __bf16 v8bf;
typedef __attribute__((ext_vector_type(8)))  float  v8f;
typedef __attribute__((ext_vector_type(4)))  unsigned int u32x4;
typedef __attribute__((ext_vector_type(8)))  int i32x8;
typedef __attribute__((ext_vector_type(4)))  int i32x4;

union FragA { v16bf v; v8bf h[2]; };
union AccF  { v8f v; float f[8]; };
union U4  { u32x4 v; unsigned int w[4]; };
union I8  { i32x8 v; int w[8]; };

static __device__ __forceinline__ v8f wmma_bf16(v16bf a, v16bf b, v8f c) {
  // 8-arg form: (neg_a, A, neg_b, B, c_mod, C, reuse_a, reuse_b)
  return __builtin_amdgcn_wmma_f32_16x16x32_bf16(false, a, false, b, (short)0, c, false, false);
}

#if __has_builtin(__builtin_amdgcn_tensor_load_to_lds)
#define HAVE_TDM 1
#endif

#if defined(HAVE_TDM)
// 2D TDM load: tile (tile1 rows x tile0 elems) of 2-byte elements from
// global (row stride stride0 elems) into LDS at byte offset lds_off,
// packed row-major. D# bitfields per CDNA5 ISA.
static __device__ __forceinline__ void tdm_load_2d_bf16(
    unsigned int lds_off, unsigned long long gaddr,
    unsigned int td0, unsigned int td1,
    unsigned int tile0, unsigned int tile1, unsigned int stride0)
{
  U4 g0;
  g0.w[0] = 1u;                                            // count=1, user mode
  g0.w[1] = lds_off;                                       // lds_addr (bytes)
  g0.w[2] = (unsigned int)(gaddr & 0xFFFFFFFFull);         // global_addr[31:0]
  g0.w[3] = (unsigned int)((gaddr >> 32) & 0x1FFFFFFull)   // global_addr[56:32]
          | (2u << 30);                                    // type = 2 ("image")
  I8 g1;
  g1.w[0] = (int)(1u << 16);                               // data_size=1 -> 2B
  g1.w[1] = (int)((td0 & 0xFFFFu) << 16);                  // tensor_dim0[15:0]
  g1.w[2] = (int)(((td0 >> 16) & 0xFFFFu) | ((td1 & 0xFFFFu) << 16));
  g1.w[3] = (int)(((td1 >> 16) & 0xFFFFu) | ((tile0 & 0xFFFFu) << 16));
  g1.w[4] = (int)(tile1 & 0xFFFFu);                        // tile_dim1 (tile_dim2=0)
  g1.w[5] = (int)stride0;                                  // tensor_dim0_stride[31:0]
  g1.w[6] = 0;                                             // stride0 hi / stride1 lo
  g1.w[7] = 0;
  i32x4 z4 = {0, 0, 0, 0};
#if defined(__clang_major__) && (__clang_major__ >= 23)
  i32x8 z8 = {0, 0, 0, 0, 0, 0, 0, 0};
  __builtin_amdgcn_tensor_load_to_lds(g0.v, g1.v, z4, z4, z8, 0);
#else
  __builtin_amdgcn_tensor_load_to_lds(g0.v, g1.v, z4, z4, 0);
#endif
}
#endif

// ---------------------------------------------------------------------------
// fp32 -> bf16 conversion (bandwidth-trivial)
// ---------------------------------------------------------------------------
__global__ __launch_bounds__(256) void cvt_f32_bf16(const float* __restrict__ src,
                                                    bf16_t* __restrict__ dst, int n) {
  int i = blockIdx.x * blockDim.x + threadIdx.x;
  int stride = gridDim.x * blockDim.x;
  for (; i < n; i += stride) dst[i] = (bf16_t)src[i];
}

// ---------------------------------------------------------------------------
// Fused QKV projection: out[m,o] = sum_h x[m,h] * W[o,h]
//   z=0 -> Q [b,h,s,d] ; z=1 -> K [b,h,s,d] ; z=2 -> V transposed [b,h,d,s]
// Workgroup: 8 waves (2M x 4N), WG tile 64(M) x 256(N); wave tile 32x64.
// ---------------------------------------------------------------------------
__global__ __launch_bounds__(256) void qkv_gemm(
    const bf16_t* __restrict__ xb,
    const bf16_t* __restrict__ wqb, const bf16_t* __restrict__ wkb,
    const bf16_t* __restrict__ wvb,
    bf16_t* __restrict__ qb, bf16_t* __restrict__ kb, bf16_t* __restrict__ vtb)
{
  const int lane = threadIdx.x & 31;
  const int wave = threadIdx.x >> 5;
  const int hi = lane >> 4, ln = lane & 15;
  const int wm = wave & 1, wn = wave >> 1;
  const int m0 = blockIdx.x * 64 + wm * 32;
  const int n0 = blockIdx.y * 256 + wn * 64;
  const int z  = blockIdx.z;
  const bf16_t* W = (z == 0) ? wqb : (z == 1) ? wkb : wvb;

  const bf16_t* arow0 = xb + (size_t)(m0 + ln) * HH;
  const bf16_t* arow1 = xb + (size_t)(m0 + 16 + ln) * HH;

  v8f zf = {0.f,0.f,0.f,0.f,0.f,0.f,0.f,0.f};
  AccF acc[2][4];
#pragma unroll
  for (int i = 0; i < 2; ++i)
#pragma unroll
    for (int t = 0; t < 4; ++t) acc[i][t].v = zf;

  for (int k = 0; k < HH; k += 32) {
    FragA af0, af1;
    af0.h[0] = *(const v8bf*)(arow0 + k + hi * 8);
    af0.h[1] = *(const v8bf*)(arow0 + k + 16 + hi * 8);
    af1.h[0] = *(const v8bf*)(arow1 + k + hi * 8);
    af1.h[1] = *(const v8bf*)(arow1 + k + 16 + hi * 8);
#pragma unroll
    for (int t = 0; t < 4; ++t) {
      const bf16_t* brow = W + (size_t)(n0 + t * 16 + ln) * HH + k + hi * 8;
      FragA bfr;
      bfr.h[0] = *(const v8bf*)(brow);
      bfr.h[1] = *(const v8bf*)(brow + 16);
      acc[0][t].v = wmma_bf16(af0.v, bfr.v, acc[0][t].v);
      acc[1][t].v = wmma_bf16(af1.v, bfr.v, acc[1][t].v);
    }
  }

#pragma unroll
  for (int i = 0; i < 2; ++i) {
#pragma unroll
    for (int t = 0; t < 4; ++t) {
      const int o = n0 + t * 16 + ln;
      const int hh = o >> 6, d = o & 63;
#pragma unroll
      for (int r = 0; r < 8; ++r) {
        const int m = m0 + i * 16 + r + hi * 8;
        const int bidx = m >> 11, s = m & (SS - 1);
        const bf16_t val = (bf16_t)acc[i][t].f[r];
        if (z == 0)
          qb[((size_t)(bidx * NH + hh) * SS + s) * HD + d] = val;
        else if (z == 1)
          kb[((size_t)(bidx * NH + hh) * SS + s) * HD + d] = val;
        else
          vtb[((size_t)(bidx * NH + hh) * HD + d) * SS + s] = val;
      }
    }
  }
}

// ---------------------------------------------------------------------------
// Flash-style attention: per wave 16 queries, stream keys in chunks of 64.
// K/V chunks staged once per WG into LDS via the Tensor Data Mover
// (fallback: cooperative copy). Online softmax with mask (-1e30);
// P goes D-layout -> LDS -> A-layout. Output merged [b, s, h*64+d] bf16.
// ---------------------------------------------------------------------------
__global__ __launch_bounds__(256) void attn_kernel(
    const bf16_t* __restrict__ qb, const bf16_t* __restrict__ kb,
    const bf16_t* __restrict__ vtb, const unsigned char* __restrict__ mask,
    bf16_t* __restrict__ mb)
{
  // [0,4096): K chunk 64keys x 64hd ; [4096,8192): V chunk 64hd x 64keys ;
  // [8192,16384): P staging, 16x64 per wave. Total 32 KB.
  __shared__ unsigned short lds_all[4096 + 4096 + 8 * 16 * 64];

  const int lane = threadIdx.x & 31;
  const int wave = threadIdx.x >> 5;
  const int hi = lane >> 4, ln = lane & 15;
  const int q0 = blockIdx.x * 128 + wave * 16;
  const int h = blockIdx.y, b = blockIdx.z;

  const bf16_t* qp = qb  + (size_t)(b * NH + h) * SS * HD;
  const bf16_t* kp = kb  + (size_t)(b * NH + h) * SS * HD;
  const bf16_t* vp = vtb + (size_t)(b * NH + h) * HD * SS;
  const unsigned char* mk = mask + (size_t)b * SS;

  bf16_t* kT = reinterpret_cast<bf16_t*>(lds_all);           // [64][64]
  bf16_t* vT = reinterpret_cast<bf16_t*>(lds_all) + 4096;    // [64][64]
  bf16_t* wp = reinterpret_cast<bf16_t*>(lds_all) + 8192 + wave * (16 * 64);

#if defined(HAVE_TDM)
  const unsigned int ldsK_off = (unsigned int)(unsigned long long)(uintptr_t)kT;
  const unsigned int ldsV_off = (unsigned int)(unsigned long long)(uintptr_t)vT;
#endif

  // Q fragments resident for the whole key sweep (K-dim = hd = 64 -> 2 frags)
  FragA qf[2];
  {
    const bf16_t* qrow = qp + (size_t)(q0 + ln) * HD;
#pragma unroll
    for (int f = 0; f < 2; ++f) {
      qf[f].h[0] = *(const v8bf*)(qrow + f * 32 + hi * 8);
      qf[f].h[1] = *(const v8bf*)(qrow + f * 32 + 16 + hi * 8);
    }
  }

  v8f zf = {0.f,0.f,0.f,0.f,0.f,0.f,0.f,0.f};
  AccF o[4];
#pragma unroll
  for (int t = 0; t < 4; ++t) o[t].v = zf;
  float mi[8], li[8];
#pragma unroll
  for (int r = 0; r < 8; ++r) { mi[r] = -1.0e30f; li[r] = 0.f; }

  for (int kc = 0; kc < SS; kc += 64) {
    // ---- stage K (64x64, row=key) and V (64x64, row=hd) chunks in LDS ----
#if defined(HAVE_TDM)
    if (wave == 0) {
      tdm_load_2d_bf16(ldsK_off,
                       (unsigned long long)(uintptr_t)(kp + (size_t)kc * HD),
                       /*td0=*/HD, /*td1=*/64, /*tile0=*/64, /*tile1=*/64,
                       /*stride0=*/HD);
      tdm_load_2d_bf16(ldsV_off,
                       (unsigned long long)(uintptr_t)(vp + (size_t)kc),
                       /*td0=*/SS, /*td1=*/64, /*tile0=*/64, /*tile1=*/64,
                       /*stride0=*/SS);
#if __has_builtin(__builtin_amdgcn_s_wait_tensorcnt)
      __builtin_amdgcn_s_wait_tensorcnt(0);
#endif
    }
#else
    {
      const int tid = threadIdx.x;
      // K chunk: contiguous 4096 bf16 -> 512 x v8bf, 2 per thread
      const v8bf* srcK = (const v8bf*)(kp + (size_t)kc * HD);
      v8bf* dstK = (v8bf*)kT;
      dstK[tid] = srcK[tid];
      dstK[tid + 256] = srcK[tid + 256];
      // V chunk: 64 rows of 64 from strided Vt
      const int vr = tid >> 2, vs = (tid & 3) * 16;
      const bf16_t* srcV = vp + (size_t)vr * SS + kc + vs;
      *(v8bf*)(vT + vr * 64 + vs)     = *(const v8bf*)(srcV);
      *(v8bf*)(vT + vr * 64 + vs + 8) = *(const v8bf*)(srcV + 8);
    }
#endif
    __syncthreads();

    // ---- scores: Q (16xhd) @ K^T (hd x 64keys), 4 N-tiles x 2 K-steps ----
    float sc[4][8];
#pragma unroll
    for (int t = 0; t < 4; ++t) {
      const bf16_t* krow = kT + (t * 16 + ln) * 64;  // B col = key row (LDS)
      FragA k0, k1;
      k0.h[0] = *(const v8bf*)(krow + hi * 8);
      k0.h[1] = *(const v8bf*)(krow + 16 + hi * 8);
      k1.h[0] = *(const v8bf*)(krow + 32 + hi * 8);
      k1.h[1] = *(const v8bf*)(krow + 48 + hi * 8);
      AccF a; a.v = zf;
      a.v = wmma_bf16(qf[0].v, k0.v, a.v);
      a.v = wmma_bf16(qf[1].v, k1.v, a.v);
      const float madd = mk[kc + t * 16 + ln] ? 0.f : -1.0e30f;
#pragma unroll
      for (int r = 0; r < 8; ++r) sc[t][r] = a.f[r] * 0.125f + madd;
    }

    // ---- online softmax (rows live in 16-lane halves; reduce width 16) ----
    float nm[8], fac[8];
#pragma unroll
    for (int r = 0; r < 8; ++r) {
      float cm = fmaxf(fmaxf(sc[0][r], sc[1][r]), fmaxf(sc[2][r], sc[3][r]));
      cm = fmaxf(cm, __shfl_xor(cm, 1, 16));
      cm = fmaxf(cm, __shfl_xor(cm, 2, 16));
      cm = fmaxf(cm, __shfl_xor(cm, 4, 16));
      cm = fmaxf(cm, __shfl_xor(cm, 8, 16));
      nm[r]  = fmaxf(mi[r], cm);
      fac[r] = __expf(mi[r] - nm[r]);
      mi[r]  = nm[r];
    }
#pragma unroll
    for (int t = 0; t < 4; ++t)
#pragma unroll
      for (int r = 0; r < 8; ++r) sc[t][r] = __expf(sc[t][r] - nm[r]);
#pragma unroll
    for (int r = 0; r < 8; ++r) {
      float rs = sc[0][r] + sc[1][r] + sc[2][r] + sc[3][r];
      rs += __shfl_xor(rs, 1, 16);
      rs += __shfl_xor(rs, 2, 16);
      rs += __shfl_xor(rs, 4, 16);
      rs += __shfl_xor(rs, 8, 16);
      li[r] = li[r] * fac[r] + rs;
    }
#pragma unroll
    for (int t = 0; t < 4; ++t)
#pragma unroll
      for (int r = 0; r < 8; ++r) o[t].f[r] *= fac[r];

    // ---- P: D-layout (row=r+8*hi, col=t*16+ln) -> LDS row-major 16x64 ----
#pragma unroll
    for (int t = 0; t < 4; ++t)
#pragma unroll
      for (int r = 0; r < 8; ++r)
        wp[(r + hi * 8) * 64 + t * 16 + ln] = (bf16_t)sc[t][r];
    __syncthreads();

    // ---- reload P as A-fragments (row = ln, two 8-elem runs per frag) ----
    FragA pf[2];
#pragma unroll
    for (int f = 0; f < 2; ++f) {
      pf[f].h[0] = *(const v8bf*)(wp + ln * 64 + f * 32 + hi * 8);
      pf[f].h[1] = *(const v8bf*)(wp + ln * 64 + f * 32 + 16 + hi * 8);
    }

    // ---- O += P (16x64keys) @ V (64keys x 64hd); V rows = hd (LDS) ----
#pragma unroll
    for (int t = 0; t < 4; ++t) {
      const bf16_t* vrow = vT + (t * 16 + ln) * 64;
#pragma unroll
      for (int f = 0; f < 2; ++f) {
        FragA vf;
        vf.h[0] = *(const v8bf*)(vrow + f * 32 + hi * 8);
        vf.h[1] = *(const v8bf*)(vrow + f * 32 + 16 + hi * 8);
        o[t].v = wmma_bf16(pf[f].v, vf.v, o[t].v);
      }
    }
    __syncthreads();   // protect K/V/P LDS before next chunk's staging
  }

  // ---- epilogue: normalize and store merged [b, s, h*64+d] ----
#pragma unroll
  for (int t = 0; t < 4; ++t) {
#pragma unroll
    for (int r = 0; r < 8; ++r) {
      const int q = q0 + r + hi * 8;
      const float l = li[r];
      const float ov = (l > 0.f) ? o[t].f[r] / l : 0.f;
      mb[((size_t)(b * SS + q)) * HH + h * HD + t * 16 + ln] = (bf16_t)ov;
    }
  }
}

// ---------------------------------------------------------------------------
// Output projection: out[m,o] = sum_h merged[m,h] * Wp[o,h], fp32 result.
// Same 32x64 wave tiling as qkv_gemm.
// ---------------------------------------------------------------------------
__global__ __launch_bounds__(256) void proj_gemm(
    const bf16_t* __restrict__ mb, const bf16_t* __restrict__ wpb,
    float* __restrict__ out)
{
  const int lane = threadIdx.x & 31;
  const int wave = threadIdx.x >> 5;
  const int hi = lane >> 4, ln = lane & 15;
  const int wm = wave & 1, wn = wave >> 1;
  const int m0 = blockIdx.x * 64 + wm * 32;
  const int n0 = blockIdx.y * 256 + wn * 64;

  const bf16_t* arow0 = mb + (size_t)(m0 + ln) * HH;
  const bf16_t* arow1 = mb + (size_t)(m0 + 16 + ln) * HH;

  v8f zf = {0.f,0.f,0.f,0.f,0.f,0.f,0.f,0.f};
  AccF acc[2][4];
#pragma unroll
  for (int i = 0; i < 2; ++i)
#pragma unroll
    for (int t = 0; t < 4; ++t) acc[i][t].v = zf;

  for (int k = 0; k < HH; k += 32) {
    FragA af0, af1;
    af0.h[0] = *(const v8bf*)(arow0 + k + hi * 8);
    af0.h[1] = *(const v8bf*)(arow0 + k + 16 + hi * 8);
    af1.h[0] = *(const v8bf*)(arow1 + k + hi * 8);
    af1.h[1] = *(const v8bf*)(arow1 + k + 16 + hi * 8);
#pragma unroll
    for (int t = 0; t < 4; ++t) {
      const bf16_t* brow = wpb + (size_t)(n0 + t * 16 + ln) * HH + k + hi * 8;
      FragA bfr;
      bfr.h[0] = *(const v8bf*)(brow);
      bfr.h[1] = *(const v8bf*)(brow + 16);
      acc[0][t].v = wmma_bf16(af0.v, bfr.v, acc[0][t].v);
      acc[1][t].v = wmma_bf16(af1.v, bfr.v, acc[1][t].v);
    }
  }

#pragma unroll
  for (int i = 0; i < 2; ++i) {
#pragma unroll
    for (int t = 0; t < 4; ++t) {
      const int oc = n0 + t * 16 + ln;
#pragma unroll
      for (int r = 0; r < 8; ++r) {
        const int m = m0 + i * 16 + r + hi * 8;
        out[(size_t)m * HH + oc] = acc[i][t].f[r];
      }
    }
  }
}

// ---------------------------------------------------------------------------
extern "C" void kernel_launch(void* const* d_in, const int* in_sizes, int n_in,
                              void* d_out, int out_size, void* d_ws, size_t ws_size,
                              hipStream_t stream) {
  (void)in_sizes; (void)n_in; (void)out_size; (void)ws_size;

  const float* x          = (const float*)d_in[0];
  const unsigned char* mk = (const unsigned char*)d_in[1];  // jnp.bool_ -> 1 byte
  const float* Wq         = (const float*)d_in[2];
  const float* Wk         = (const float*)d_in[3];
  const float* Wv         = (const float*)d_in[4];
  const float* Wp         = (const float*)d_in[5];

  char* ws = (char*)d_ws;
  size_t off = 0;
  auto take = [&](size_t bytes) -> char* {
    char* p = ws + off;
    off += (bytes + 255) & ~(size_t)255;
    return p;
  };

  const size_t nX = (size_t)BB * SS * HH;   // 3,145,728
  const size_t nW = (size_t)HH * HH;        //   589,824
  bf16_t* xb  = (bf16_t*)take(nX * 2);
  bf16_t* wqb = (bf16_t*)take(nW * 2);
  bf16_t* wkb = (bf16_t*)take(nW * 2);
  bf16_t* wvb = (bf16_t*)take(nW * 2);
  bf16_t* wpb = (bf16_t*)take(nW * 2);
  bf16_t* qb  = (bf16_t*)take(nX * 2);
  bf16_t* kb  = (bf16_t*)take(nX * 2);
  bf16_t* vtb = (bf16_t*)take(nX * 2);
  bf16_t* mb  = (bf16_t*)take(nX * 2);

  cvt_f32_bf16<<<2048, 256, 0, stream>>>(x,  xb,  (int)nX);
  cvt_f32_bf16<<<1024, 256, 0, stream>>>(Wq, wqb, (int)nW);
  cvt_f32_bf16<<<1024, 256, 0, stream>>>(Wk, wkb, (int)nW);
  cvt_f32_bf16<<<1024, 256, 0, stream>>>(Wv, wvb, (int)nW);
  cvt_f32_bf16<<<1024, 256, 0, stream>>>(Wp, wpb, (int)nW);

  qkv_gemm<<<dim3(64, 3, 3), 256, 0, stream>>>(xb, wqb, wkb, wvb, qb, kb, vtb);
  attn_kernel<<<dim3(SS / 128, NH, BB), 256, 0, stream>>>(qb, kb, vtb, mk, mb);
  proj_gemm<<<dim3(64, 3), 256, 0, stream>>>(mb, wpb, (float*)d_out);
}
